// simple_LSTM_cell_56401510531656
// MI455X (gfx1250) — compile-verified, matrix operand order checked by reading, not run on hardware
//
#include <hip/hip_runtime.h>

// ---------------------------------------------------------------------------
// LSTM cell + vocab projection for gfx1250 (MI455X), bf16 WMMA path.
//   c_t, h_t : [8192, 256] fp32
//   y_t      : [8192, 32000] fp32  (memory-bound: 1.05 GB out => ~45us floor)
// Projection stages shared Wl tiles into LDS via GLOBAL_LOAD_ASYNC_TO_LDS
// (ASYNCcnt-tracked, double-buffered) to kill the 8x redundant B loads.
// ---------------------------------------------------------------------------

typedef __bf16 bf16_t;
typedef __bf16 v16bf __attribute__((ext_vector_type(16)));
typedef __bf16 v8bf  __attribute__((ext_vector_type(8)));
typedef float  v8f   __attribute__((ext_vector_type(8)));

#define EMBED 256
#define HID   256
#define VOCAB 32000
#define BATCH 8192
#define KIN   (EMBED + 2 * HID)   // 768  (gate GEMM K)
#define KC    (EMBED + HID)       // 512  (candidate GEMM K)

// LDS staging for the projection kernel: 64 rows x 32 bf16 (64B) per stage,
// rows padded to 80B so lane n's b128 read hits bank (n*20)%64 (conflict-free).
#define BROW_PAD 80
#define BSTAGE   (64 * BROW_PAD)  // 5120 B per stage, double buffered

// fp32 -> bf16, round-to-nearest-even
__device__ __forceinline__ bf16_t f2bf(float f) {
  unsigned u = __builtin_bit_cast(unsigned, f);
  unsigned r = u + 0x7FFFu + ((u >> 16) & 1u);
  unsigned short h = (unsigned short)(r >> 16);
  return __builtin_bit_cast(bf16_t, h);
}

__device__ __forceinline__ v16bf mk16(v8bf lo, v8bf hi) {
  v16bf r;
#pragma unroll
  for (int i = 0; i < 8; ++i) { r[i] = lo[i]; r[i + 8] = hi[i]; }
  return r;
}

// Load one 16x32 bf16 operand tile (A or B) per the gfx1250 16-bit WMMA
// layout: lane L (m = L&15, s = L>>4) holds K = kb+s*8..+7 then kb+16+s*8..+7.
__device__ __forceinline__ v16bf load_tile(const bf16_t* __restrict__ base,
                                           int row_stride, int rowbase,
                                           int kb, int lane) {
  int rn = lane & 15, s = lane >> 4;
  const bf16_t* p = base + (size_t)(rowbase + rn) * row_stride + kb + s * 8;
  v8bf lo = *(const v8bf*)(p);
  v8bf hi = *(const v8bf*)(p + 16);
  return mk16(lo, hi);
}

__device__ __forceinline__ float sigmoidf_(float x) {
  return 1.0f / (1.0f + __expf(-x));
}

// ---------------------------------------------------------------------------
// Prep kernels: fp32 -> bf16 weight conversion, lstm_in packing
// ---------------------------------------------------------------------------
__global__ void cvt_f32_bf16(const float* __restrict__ s,
                             bf16_t* __restrict__ d, int n) {
  for (int i = blockIdx.x * blockDim.x + threadIdx.x; i < n;
       i += gridDim.x * blockDim.x)
    d[i] = f2bf(s[i]);
}

// lstm_in = [c_prev | h_prev | x] -> bf16 [BATCH, 768]
__global__ void pack_lstmin(const float* __restrict__ c,
                            const float* __restrict__ h,
                            const float* __restrict__ x,
                            bf16_t* __restrict__ d) {
  const int total = BATCH * KIN;
  for (int i = blockIdx.x * blockDim.x + threadIdx.x; i < total;
       i += gridDim.x * blockDim.x) {
    int row = i / KIN, col = i - row * KIN;
    float v;
    if (col < HID)            v = c[(size_t)row * HID + col];
    else if (col < 2 * HID)   v = h[(size_t)row * HID + (col - HID)];
    else                      v = x[(size_t)row * EMBED + (col - 2 * HID)];
    d[i] = f2bf(v);
  }
}

// ---------------------------------------------------------------------------
// Gate kernel: one wave computes a 16(batch) x 16(hid) tile of all 4 gates.
// K-loop over 768 (f/u/o) with candidate gate sharing A tiles for k >= 256.
// ---------------------------------------------------------------------------
__global__ __launch_bounds__(128) void lstm_gates_wmma(
    const bf16_t* __restrict__ lin, const bf16_t* __restrict__ wf,
    const bf16_t* __restrict__ wu,  const bf16_t* __restrict__ wo,
    const bf16_t* __restrict__ wc,
    const float* __restrict__ bfv, const float* __restrict__ buv,
    const float* __restrict__ bov, const float* __restrict__ bcv,
    const float* __restrict__ c_prev,
    float* __restrict__ c_out, float* __restrict__ h_out,
    bf16_t* __restrict__ h_bf) {
  int lane = threadIdx.x & 31;
  int wave = threadIdx.x >> 5;
  int tile = blockIdx.x * 4 + wave;       // 0..8191 tiles (512 bm x 16 bn)
  int bm   = (tile >> 4) * 16;            // batch row base
  int bn   = (tile & 15) * 16;            // hidden col base

  v8f aF = {}, aU = {}, aO = {}, aC = {};
  for (int kb = 0; kb < KIN; kb += 32) {
    v16bf A  = load_tile(lin, KIN, bm, kb, lane);
    v16bf Bf = load_tile(wf,  KIN, bn, kb, lane);
    v16bf Bu = load_tile(wu,  KIN, bn, kb, lane);
    v16bf Bo = load_tile(wo,  KIN, bn, kb, lane);
    aF = __builtin_amdgcn_wmma_f32_16x16x32_bf16(false, A, false, Bf,
                                                 (short)0, aF, false, false);
    aU = __builtin_amdgcn_wmma_f32_16x16x32_bf16(false, A, false, Bu,
                                                 (short)0, aU, false, false);
    aO = __builtin_amdgcn_wmma_f32_16x16x32_bf16(false, A, false, Bo,
                                                 (short)0, aO, false, false);
    if (kb >= HID) {  // candidate gate input is lstm_in[:, 256:768]
      v16bf Bc = load_tile(wc, KC, bn, kb - HID, lane);
      aC = __builtin_amdgcn_wmma_f32_16x16x32_bf16(false, A, false, Bc,
                                                   (short)0, aC, false, false);
    }
  }

  // C/D layout: VGPR r holds M = r + 8*(lane>>4), N = lane & 15
  int n = lane & 15, s = lane >> 4;
  float bF = bfv[bn + n], bU = buv[bn + n], bO = bov[bn + n], bC = bcv[bn + n];
#pragma unroll
  for (int r = 0; r < 8; ++r) {
    int row = bm + r + 8 * s;
    int col = bn + n;
    size_t idx = (size_t)row * HID + col;
    float fg = sigmoidf_(aF[r] + bF);
    float ug = sigmoidf_(aU[r] + bU);
    float og = sigmoidf_(aO[r] + bO);
    float cn = tanhf(aC[r] + bC);
    float cp = c_prev[idx];
    float ct = fg * cp + ug * cn;
    float ht = og * tanhf(ct);
    c_out[idx] = ct;
    h_out[idx] = ht;
    h_bf[idx]  = f2bf(ht);
  }
}

// ---------------------------------------------------------------------------
// Vocab projection: y = h @ Wl.T + bl   [8192 x 32000]
// Block = 256 threads (8 waves), macro tile 128(M) x 64(N).
// The 64x32 Wl tile (4KB) shared by all 8 waves is staged into LDS once per
// K-step via GLOBAL_LOAD_ASYNC_TO_LDS_B128 (each lane moves 16B), double
// buffered so the next stage's fill overlaps the current WMMAs.
// ---------------------------------------------------------------------------
__global__ __launch_bounds__(256) void vocab_proj_wmma(
    const bf16_t* __restrict__ h_bf, const bf16_t* __restrict__ wl,
    const float* __restrict__ bl, float* __restrict__ y) {
  __shared__ __align__(16) unsigned char smem[2 * BSTAGE];

  int lane = threadIdx.x & 31;
  int wave = threadIdx.x >> 5;
  int bm = blockIdx.y * 128 + wave * 16;
  int bn = blockIdx.x * 64;

  // Per-thread async-copy slot: row = tid/4 (0..63), 16B chunk = tid%4.
  const int crow = threadIdx.x >> 2;
  const int cchk = threadIdx.x & 3;
  const unsigned lds_base = (unsigned)(uintptr_t)(void*)smem;  // low 32 = LDS off
  const bf16_t* gsrc_row = wl + (size_t)(bn + crow) * HID + cchk * 8;

  // Stage kb-th 64x32 Wl tile into LDS buffer `buf` (one b128 per lane).
  auto issue_stage = [&](int kb, int buf) {
    unsigned ldsoff = lds_base + buf * BSTAGE + crow * BROW_PAD + cchk * 16;
    const bf16_t* g = gsrc_row + kb;
    asm volatile("global_load_async_to_lds_b128 %0, %1, off"
                 :: "v"(ldsoff), "v"(g)
                 : "memory");
  };

  issue_stage(0, 0);

  v8f a0 = {}, a1 = {}, a2 = {}, a3 = {};
  const int n = lane & 15, s = lane >> 4;

#pragma unroll
  for (int i = 0; i < HID / 32; ++i) {   // 8 K-steps
    const int kb = i * 32;
    const int buf = i & 1;
    if (i + 1 < HID / 32) {
      issue_stage(kb + 32, buf ^ 1);     // overlap next fill with compute
      asm volatile("s_wait_asynccnt 0x1" ::: "memory");
    } else {
      asm volatile("s_wait_asynccnt 0x0" ::: "memory");
    }
    __syncthreads();                     // all lanes' chunks landed in LDS

    v16bf A = load_tile(h_bf, HID, bm, kb, lane);

    const unsigned char* sb = smem + buf * BSTAGE;
#pragma unroll
    for (int t = 0; t < 4; ++t) {
      const unsigned char* rp = sb + (t * 16 + n) * BROW_PAD + s * 16;
      v8bf lo = *(const v8bf*)(rp);
      v8bf hi = *(const v8bf*)(rp + 32);
      v16bf B = mk16(lo, hi);
      v8f* acc = (t == 0) ? &a0 : (t == 1) ? &a1 : (t == 2) ? &a2 : &a3;
      *acc = __builtin_amdgcn_wmma_f32_16x16x32_bf16(false, A, false, B,
                                                     (short)0, *acc, false,
                                                     false);
    }
    __syncthreads();                     // reads done before buf is refilled
  }

#pragma unroll
  for (int r = 0; r < 8; ++r) {
    int row = bm + r + 8 * s;
    size_t base = (size_t)row * VOCAB + bn;
    y[base +  0 + n] = a0[r] + bl[bn +  0 + n];
    y[base + 16 + n] = a1[r] + bl[bn + 16 + n];
    y[base + 32 + n] = a2[r] + bl[bn + 32 + n];
    y[base + 48 + n] = a3[r] + bl[bn + 48 + n];
  }
}

// ---------------------------------------------------------------------------
// Launch
// ---------------------------------------------------------------------------
extern "C" void kernel_launch(void* const* d_in, const int* in_sizes, int n_in,
                              void* d_out, int out_size, void* d_ws,
                              size_t ws_size, hipStream_t stream) {
  const float* x      = (const float*)d_in[0];
  const float* c_prev = (const float*)d_in[1];
  const float* h_prev = (const float*)d_in[2];
  const float* Wf = (const float*)d_in[3];  const float* bfb = (const float*)d_in[4];
  const float* Wu = (const float*)d_in[5];  const float* bub = (const float*)d_in[6];
  const float* Wo = (const float*)d_in[7];  const float* bob = (const float*)d_in[8];
  const float* Wc = (const float*)d_in[9];  const float* bcb = (const float*)d_in[10];
  const float* Wl = (const float*)d_in[11]; const float* blb = (const float*)d_in[12];

  // Workspace layout (bf16 elements, all 16B-aligned): ~34.6 MB total
  bf16_t* ws  = (bf16_t*)d_ws;
  bf16_t* lin = ws;                                  // BATCH*KIN   = 6,291,456
  bf16_t* wfb = lin + (size_t)BATCH * KIN;           // HID*KIN     =   196,608
  bf16_t* wub = wfb + (size_t)HID * KIN;
  bf16_t* wob = wub + (size_t)HID * KIN;
  bf16_t* wcb = wob + (size_t)HID * KIN;             // HID*KC      =   131,072
  bf16_t* wlb = wcb + (size_t)HID * KC;              // VOCAB*HID   = 8,192,000
  bf16_t* hbf = wlb + (size_t)VOCAB * HID;           // BATCH*HID   = 2,097,152

  float* c_out = (float*)d_out;                      // [BATCH, HID]
  float* h_out = c_out + (size_t)BATCH * HID;        // [BATCH, HID]
  float* y_out = h_out + (size_t)BATCH * HID;        // [BATCH, VOCAB]

  // 1) precision conversion / packing
  pack_lstmin<<<2048, 256, 0, stream>>>(c_prev, h_prev, x, lin);
  cvt_f32_bf16<<<768, 256, 0, stream>>>(Wf, wfb, HID * KIN);
  cvt_f32_bf16<<<768, 256, 0, stream>>>(Wu, wub, HID * KIN);
  cvt_f32_bf16<<<768, 256, 0, stream>>>(Wo, wob, HID * KIN);
  cvt_f32_bf16<<<512, 256, 0, stream>>>(Wc, wcb, HID * KC);
  cvt_f32_bf16<<<8192, 256, 0, stream>>>(Wl, wlb, VOCAB * HID);

  // 2) fused gates -> c_t, h_t (fp32) + h_t (bf16)
  lstm_gates_wmma<<<2048, 128, 0, stream>>>(lin, wfb, wub, wob, wcb, bfb, bub,
                                            bob, bcb, c_prev, c_out, h_out,
                                            hbf);

  // 3) vocab projection -> y_t
  vocab_proj_wmma<<<dim3(500, 64), 256, 0, stream>>>(hbf, wlb, blb, y_out);
}
